// AutoregressiveDecoder_74285754351936
// MI455X (gfx1250) — matrix operand order repaired and checked by reading, not compiled, not run on hardware
//
#include <hip/hip_runtime.h>
#include <hip/hip_bf16.h>
#include <math.h>

typedef __bf16 bf16_t;
typedef bf16_t v16bf __attribute__((ext_vector_type(16)));
typedef bf16_t v8bf  __attribute__((ext_vector_type(8)));
typedef float  v8f   __attribute__((ext_vector_type(8)));

#define B_   64
#define L_   128
#define E_   512
#define H_   512
#define V_   32000
#define T_   64
#define K3H  (3 * H_)   // 1536

__device__ __forceinline__ unsigned short f32_to_bf16(float f) {
  unsigned u = __float_as_uint(f);
  u += 0x7FFFu + ((u >> 16) & 1u);           // round-to-nearest-even
  return (unsigned short)(u >> 16);
}

// ---------------- weight fp32 -> bf16 conversion ----------------
__global__ void k_convert(const float* __restrict__ src,
                          unsigned short* __restrict__ dst, int n) {
  int i = blockIdx.x * blockDim.x + threadIdx.x;
  int stride = gridDim.x * blockDim.x;
  for (; i < n; i += stride) dst[i] = f32_to_bf16(src[i]);
}

// ---------------- h0 = z @ W_proj^T + b_proj (tiny GEMM, K=128) ----------------
__global__ void k_init(const float* __restrict__ z, const float* __restrict__ Wp,
                       const float* __restrict__ bp,
                       float* __restrict__ h, unsigned short* __restrict__ hb,
                       unsigned short* __restrict__ xb,
                       unsigned long long* __restrict__ keys) {
  int i = blockIdx.x * blockDim.x + threadIdx.x;
  if (i >= B_ * H_) return;
  int b = i / H_, j = i % H_;
  const float* zr = z + b * L_;
  const float* wr = Wp + j * L_;
  float acc = bp[j];
#pragma unroll 8
  for (int l = 0; l < L_; ++l) acc += zr[l] * wr[l];
  h[i] = acc;
  unsigned short bf = f32_to_bf16(acc);
  hb[i] = bf;
  xb[i] = bf;                 // first step uses x = h0
  if (i < B_) keys[i] = 0ull;
}

// ---------------- gates GEMM: G[0]=xb@Wih^T, G[1]=hb@Whh^T (WMMA bf16) ----------------
// 2 matrices * (4 m-tiles * 96 n-tiles) = 768 wave-jobs, K=512 -> 16 wmma each
__global__ void k_gates(const unsigned short* __restrict__ xb,
                        const unsigned short* __restrict__ hb,
                        const unsigned short* __restrict__ Wih,
                        const unsigned short* __restrict__ Whh,
                        float* __restrict__ G) {
  int wave = (int)((blockIdx.x * blockDim.x + threadIdx.x) >> 5);
  int lane = threadIdx.x & 31;
  int g  = wave / 384;            // 0: input gates, 1: hidden gates
  int t  = wave % 384;
  int mt = t / 96, nt = t % 96;
  const unsigned short* A = g ? hb : xb;
  const unsigned short* W = g ? Whh : Wih;
  int m = lane & 15, half = lane >> 4;
  const unsigned short* arow = A + (mt * 16 + m) * 512;
  const unsigned short* brow = W + (nt * 16 + m) * 512 + 16 * half;  // col n == lane&15
  v8f c = {};
#pragma unroll
  for (int ks = 0; ks < 16; ++ks) {
    int k0 = ks * 32;
    v16bf a, bf;
    a = *(const v16bf*)(arow + k0 + 16 * half);                   // K = k0+16h .. +15
    bf = *(const v16bf*)(brow + k0);                              // K = k0+16h .. +15
    c = __builtin_amdgcn_wmma_f32_16x16x32_bf16(false, a, false, bf,
                                                (short)0, c, false, false);
  }
  float* Gg = G + g * (B_ * K3H);
  int col = nt * 16 + (lane & 15);
#pragma unroll
  for (int r = 0; r < 8; ++r) {
    int row = mt * 16 + r + 8 * half;
    Gg[row * K3H + col] = c[r];
  }
}

// ---------------- GRU nonlinearity + h update; also resets argmax keys ----------------
__global__ void k_hupdate(const float* __restrict__ G,
                          const float* __restrict__ b_ih, const float* __restrict__ b_hh,
                          float* __restrict__ h, unsigned short* __restrict__ hb,
                          unsigned long long* __restrict__ keys) {
  int i = blockIdx.x * blockDim.x + threadIdx.x;
  if (i >= B_ * H_) return;
  int b = i / H_, j = i % H_;
  const float* gi = G + b * K3H;
  const float* gh = G + B_ * K3H + b * K3H;
  float ir = gi[j]          + b_ih[j];
  float iz = gi[H_ + j]     + b_ih[H_ + j];
  float in = gi[2 * H_ + j] + b_ih[2 * H_ + j];
  float hr = gh[j]          + b_hh[j];
  float hz = gh[H_ + j]     + b_hh[H_ + j];
  float hn = gh[2 * H_ + j] + b_hh[2 * H_ + j];
  float r  = 1.f / (1.f + __expf(-(ir + hr)));
  float zg = 1.f / (1.f + __expf(-(iz + hz)));
  float n  = tanhf(in + r * hn);
  float hv = (1.f - zg) * n + zg * h[i];
  h[i]  = hv;
  hb[i] = f32_to_bf16(hv);
  if (i < B_) keys[i] = 0ull;   // reset argmax keys for this step
}

// ---------------- logits = hb @ Wfc^T + b_fc, fused argmax ----------------
// wave-job: one 16-row m-tile x 8 consecutive 16-col n-tiles; A frags reused.
__global__ void k_logits(const unsigned short* __restrict__ hb,
                         const unsigned short* __restrict__ Wfc,
                         const float* __restrict__ b_fc,
                         float* __restrict__ out,   // pre-offset by t*V_
                         unsigned long long* __restrict__ keys) {
  const int NTPW = 8;
  const int NGRP = (V_ / 16) / NTPW;   // 250
  int wave = (int)((blockIdx.x * blockDim.x + threadIdx.x) >> 5);
  int lane = threadIdx.x & 31;
  int mt = wave / NGRP;
  int ng = wave % NGRP;
  int m = lane & 15, half = lane >> 4;

  const unsigned short* arow = hb + (mt * 16 + m) * 512;
  v16bf afrag[16];
#pragma unroll
  for (int ks = 0; ks < 16; ++ks) {
    int k0 = ks * 32;
    afrag[ks] = *(const v16bf*)(arow + k0 + 16 * half);
  }

  float bestv[8];
  unsigned besti[8];
#pragma unroll
  for (int r = 0; r < 8; ++r) { bestv[r] = -__builtin_inff(); besti[r] = 0u; }

  int col = lane & 15;
#pragma unroll 1
  for (int nt = 0; nt < NTPW; ++nt) {
    int v0 = (ng * NTPW + nt) * 16;
    const unsigned short* brow = Wfc + (v0 + col) * 512 + 16 * half;
    v8f c = {};
#pragma unroll
    for (int ks = 0; ks < 16; ++ks) {
      v16bf bf = *(const v16bf*)(brow + ks * 32);
      c = __builtin_amdgcn_wmma_f32_16x16x32_bf16(false, afrag[ks], false, bf,
                                                  (short)0, c, false, false);
    }
    float bias = b_fc[v0 + col];
#pragma unroll
    for (int r = 0; r < 8; ++r) {
      int row = mt * 16 + r + 8 * half;
      float val = c[r] + bias;
      out[(size_t)row * (T_ * V_) + v0 + col] = val;
      if (val > bestv[r]) { bestv[r] = val; besti[r] = (unsigned)(v0 + col); }
    }
  }

  // reduce (val, idx) across the 16-lane column group; first index wins ties
#pragma unroll
  for (int r = 0; r < 8; ++r) {
    unsigned u = __float_as_uint(bestv[r]);
    u = (u & 0x80000000u) ? ~u : (u | 0x80000000u);
    unsigned long long key =
        ((unsigned long long)u << 32) | (unsigned long long)(0xFFFFFFFFu - besti[r]);
#pragma unroll
    for (int s = 8; s >= 1; s >>= 1) {
      unsigned long long o = __shfl_xor(key, s, 32);
      if (o > key) key = o;
    }
    if ((lane & 15) == 0) {
      int row = mt * 16 + r + 8 * half;
      atomicMax(keys + row, key);
    }
  }
}

// ---------------- x_next = bf16(emb[argmax ids]) ----------------
__global__ void k_gather(const unsigned long long* __restrict__ keys,
                         const float* __restrict__ emb,
                         unsigned short* __restrict__ xb) {
  int i = blockIdx.x * blockDim.x + threadIdx.x;
  if (i >= B_ * E_) return;
  int b = i / E_, e = i % E_;
  unsigned id = 0xFFFFFFFFu - (unsigned)(keys[b] & 0xFFFFFFFFull);
  xb[i] = f32_to_bf16(emb[(size_t)id * E_ + e]);
}

extern "C" void kernel_launch(void* const* d_in, const int* in_sizes, int n_in,
                              void* d_out, int out_size, void* d_ws, size_t ws_size,
                              hipStream_t stream) {
  (void)in_sizes; (void)n_in; (void)out_size; (void)ws_size;
  const float* z      = (const float*)d_in[0];
  const float* emb    = (const float*)d_in[1];
  const float* W_proj = (const float*)d_in[2];
  const float* b_proj = (const float*)d_in[3];
  const float* W_ih   = (const float*)d_in[4];
  const float* b_ih   = (const float*)d_in[5];
  const float* W_hh   = (const float*)d_in[6];
  const float* b_hh   = (const float*)d_in[7];
  const float* W_fc   = (const float*)d_in[8];
  const float* b_fc   = (const float*)d_in[9];
  float* out = (float*)d_out;

  // workspace layout (all chunks 256B-aligned by construction)
  char* ws = (char*)d_ws;
  unsigned short* Wfc_bf = (unsigned short*)ws;            ws += (size_t)V_ * H_ * 2;   // 32.77 MB
  unsigned short* Wih_bf = (unsigned short*)ws;            ws += (size_t)K3H * E_ * 2;  // 1.57 MB
  unsigned short* Whh_bf = (unsigned short*)ws;            ws += (size_t)K3H * H_ * 2;  // 1.57 MB
  float*          h      = (float*)ws;                     ws += (size_t)B_ * H_ * 4;
  unsigned short* hb     = (unsigned short*)ws;            ws += (size_t)B_ * H_ * 2;
  unsigned short* xb     = (unsigned short*)ws;            ws += (size_t)B_ * E_ * 2;
  float*          G      = (float*)ws;                     ws += (size_t)2 * B_ * K3H * 4;
  unsigned long long* keys = (unsigned long long*)ws;      ws += (size_t)B_ * 8;

  k_convert<<<2048, 256, 0, stream>>>(W_fc, Wfc_bf, V_ * H_);
  k_convert<<<512,  256, 0, stream>>>(W_ih, Wih_bf, K3H * E_);
  k_convert<<<512,  256, 0, stream>>>(W_hh, Whh_bf, K3H * H_);
  k_init<<<(B_ * H_ + 255) / 256, 256, 0, stream>>>(z, W_proj, b_proj, h, hb, xb, keys);

  for (int t = 0; t < T_; ++t) {
    k_gates<<<96, 256, 0, stream>>>(xb, hb, Wih_bf, Whh_bf, G);      // 768 wave-jobs
    k_hupdate<<<(B_ * H_ + 255) / 256, 256, 0, stream>>>(G, b_ih, b_hh, h, hb, keys);
    k_logits<<<125, 256, 0, stream>>>(hb, Wfc_bf, b_fc, out + (size_t)t * V_, keys); // 1000 wave-jobs
    if (t + 1 < T_)
      k_gather<<<(B_ * E_ + 255) / 256, 256, 0, stream>>>(keys, emb, xb);
  }
}
